// RNNModel_43482248905096
// MI455X (gfx1250) — compile-verified, matrix operand order checked by reading, not run on hardware
//
#include <hip/hip_runtime.h>
#include <hip/hip_bf16.h>
#include <math.h>
#include <stdint.h>

// ---------------- model dims ----------------
#define VOCAB 32000
#define EMB   1024
#define HID   1024
#define BATCH 16
#define TLEN  512
#define ROWS  (BATCH * TLEN)       // 8192

typedef __attribute__((ext_vector_type(16))) __bf16 v16bf;
typedef __attribute__((ext_vector_type(8)))  __bf16 v8bf;
typedef __attribute__((ext_vector_type(8)))  float  v8f;

// ---- fragment loader: CDNA5 16-bit A/B layout ----
// lane half 0 (lanes 0-15):  K = {0..7} U {16..23}
// lane half 1 (lanes 16-31): K = {8..15} U {24..31}
// p must already point at (row base + k0 + laneHalf), 16B aligned.
__device__ __forceinline__ v16bf frag_from(const __bf16* p) {
  v8bf lo = *(const v8bf*)(p);
  v8bf hi = *(const v8bf*)(p + 16);
  v16bf r;
#pragma unroll
  for (int i = 0; i < 8; ++i) { r[i] = lo[i]; r[i + 8] = hi[i]; }
  return r;
}

// ---- async global->LDS staging (CDNA5), with safe fallback ----
#if __has_builtin(__builtin_amdgcn_global_load_async_to_lds_b128)
#define HAS_ASYNC_LDS 1
#else
#define HAS_ASYNC_LDS 0
#endif

typedef int v4i __attribute__((vector_size(16)));
typedef __attribute__((address_space(1))) v4i v4i_glob;   // HIP AS1 ("__device__")
typedef __attribute__((address_space(3))) v4i v4i_lds;    // HIP AS3 ("__shared__")

__device__ __forceinline__ void stage_b128(const __bf16* g, __bf16* l) {
#if HAS_ASYNC_LDS
  __builtin_amdgcn_global_load_async_to_lds_b128(
      (v4i_glob*)(g), (v4i_lds*)(l), /*offset=*/0, /*cpol=*/0);
#else
  *(v8bf*)l = *(const v8bf*)g;   // sync VGPR round-trip fallback
#endif
}
__device__ __forceinline__ void stage32B(const __bf16* g, __bf16* l) {
  stage_b128(g, l);
  stage_b128(g + 8, l + 8);
}
__device__ __forceinline__ void wait_async_le2() {
#if __has_builtin(__builtin_amdgcn_s_wait_asynccnt)
  __builtin_amdgcn_s_wait_asynccnt(2);
#else
  asm volatile("s_wait_asynccnt 0x2" ::: "memory");
#endif
}
__device__ __forceinline__ void wait_async_le0() {
#if __has_builtin(__builtin_amdgcn_s_wait_asynccnt)
  __builtin_amdgcn_s_wait_asynccnt(0);
#else
  asm volatile("s_wait_asynccnt 0x0" ::: "memory");
#endif
}

// ---------------- generic bf16 WMMA GEMM ----------------
// C[M,N] (f32) = A[M,K](bf16,row-major) * W[N,K](bf16,row-major)^T + bias1 + bias2
// block = 256 threads = 8 waves; wave tile = 16(M) x 64(N); block tile = 128 x 64.
// B tile (64 rows x 64 K) is double-buffered in LDS, filled with async
// global->LDS loads; last K iteration is peeled so the hot loop is branch-free.
#define KSTEP    64
#define KSTRIDE  72   // 64 + 8 bf16 pad: 144B row stride (16B aligned, bank-skewed)

__global__ void __launch_bounds__(256)
wmma_gemm_bf16(const __bf16* __restrict__ A,
               const __bf16* __restrict__ W,
               float* __restrict__ C,
               const float* __restrict__ bias1,
               const float* __restrict__ bias2,
               int M, int N, int K)
{
  __shared__ __bf16 wtile[2][64 * KSTRIDE];   // 2 x 9216 B
  const int tid      = threadIdx.x;
  const int lane     = tid & 31;
  const int wave     = tid >> 5;
  const int laneHalf = (lane >> 4) << 3;      // 0 or 8
  const int m0 = blockIdx.x * 128 + wave * 16;
  const int n0 = blockIdx.y * 64;

  v8f acc[4] = {};

  // staging map: thread -> (row nl, 16-element K chunk kch) of the 64x64 tile
  const int nl  = tid >> 2;                   // 0..63
  const int kch = (tid & 3) << 4;             // 0,16,32,48
  const __bf16* arow = A + (size_t)(m0 + (lane & 15)) * K;
  const __bf16* wrow = W + (size_t)(n0 + nl) * K + kch;
  __bf16* l0 = &wtile[0][nl * KSTRIDE + kch];
  __bf16* l1 = &wtile[1][nl * KSTRIDE + kch];

  // one 64-K tile worth of WMMAs from LDS buffer `tile`
  auto compute_tile = [&](const __bf16* tile, int k0) {
#pragma unroll
    for (int kc = 0; kc < 2; ++kc) {          // 2 x k32 chunks
      v16bf a = frag_from(arow + k0 + kc * 32 + laneHalf);
#pragma unroll
      for (int j = 0; j < 4; ++j) {
        v16bf b = frag_from(&tile[(j * 16 + (lane & 15)) * KSTRIDE + kc * 32 + laneHalf]);
        acc[j] = __builtin_amdgcn_wmma_f32_16x16x32_bf16(
            false, a, false, b, (short)0, acc[j], false, false);
      }
    }
  };

  const int niter = K / KSTEP;
  stage32B(wrow, l0);                          // prime buffer 0

  // steady state: branch-free software pipeline
  for (int i = 0; i < niter - 1; ++i) {
    const int k0 = i * KSTEP;
    stage32B(wrow + k0 + KSTEP, (i & 1) ? l0 : l1);   // issue next tile
    wait_async_le2();                                  // current tile landed (mine)
    __syncthreads();                                   // ... for every wave
    __builtin_prefetch(arow + k0 + KSTEP, 0, 3);
    compute_tile(wtile[i & 1], k0);
    __syncthreads();                                   // reads done before re-stage
  }

  // peeled last iteration: no staging, drain to zero
  wait_async_le0();
  __syncthreads();
  compute_tile(wtile[(niter - 1) & 1], (niter - 1) * KSTEP);

#pragma unroll
  for (int j = 0; j < 4; ++j) {
    const int n = n0 + j * 16 + (lane & 15);
    float badd = (bias1 ? bias1[n] : 0.f) + (bias2 ? bias2[n] : 0.f);
#pragma unroll
    for (int r = 0; r < 8; ++r) {
      const int m = m0 + r + laneHalf;        // C layout: M = r + 8*(lane>=16)
      __builtin_nontemporal_store(acc[j][r] + badd, &C[(size_t)m * N + n]);
    }
  }
}

// ---------------- persistent recurrent scan ----------------
// h_{t} = tanh(pre[:,t,:] + h_{t-1} @ Whh^T), h in LDS (bf16), B=16 == WMMA M.
// 512 threads = 16 waves; wave w owns hidden features [64w, 64w+64).
#define HPAD 1040  // 1024 + 16 bf16 pad (2080B row stride)

__global__ void __launch_bounds__(512)
rnn_scan(const float* __restrict__ pre,     // [B,T,H] f32
         const __bf16* __restrict__ Whh,    // [H,H] bf16
         __bf16* __restrict__ y,            // [B,T,H] bf16 out
         float* __restrict__ hT)            // [B,H] f32 out (final hidden)
{
  __shared__ __bf16 hbuf[BATCH * HPAD];
  const int tid      = threadIdx.x;
  const int lane     = tid & 31;
  const int wave     = tid >> 5;            // 0..15
  const int laneHalf = (lane >> 4) << 3;    // 0 or 8
  const int n0   = wave * 64;
  const int mrow = lane & 15;               // A-frag row (batch), also B-frag col

  for (int i = tid; i < BATCH * HPAD; i += 512) hbuf[i] = (__bf16)0.f;
  __syncthreads();

  for (int t = 0; t < TLEN; ++t) {
    v8f acc[4] = {};
    for (int k0 = 0; k0 < HID; k0 += 32) {
      v16bf a = frag_from(&hbuf[mrow * HPAD + k0 + laneHalf]);   // h from LDS
#pragma unroll
      for (int j = 0; j < 4; ++j) {
        const __bf16* bp = Whh + (size_t)(n0 + j * 16 + mrow) * HID + k0 + laneHalf;
        v16bf b = frag_from(bp);                                 // Whh from L2
        acc[j] = __builtin_amdgcn_wmma_f32_16x16x32_bf16(
            false, a, false, b, (short)0, acc[j], false, false);
      }
    }
    __syncthreads();   // everyone done reading h_{t-1}
#pragma unroll
    for (int j = 0; j < 4; ++j) {
      const int n = n0 + j * 16 + mrow;
#pragma unroll
      for (int r = 0; r < 8; ++r) {
        const int b = r + laneHalf;                               // batch 0..15
        float v = tanhf(acc[j][r] + pre[((size_t)b * TLEN + t) * HID + n]);
        __bf16 vb = (__bf16)v;
        hbuf[b * HPAD + n] = vb;
        y[((size_t)b * TLEN + t) * HID + n] = vb;
        if (t == TLEN - 1) hT[b * HID + n] = v;
      }
    }
    __syncthreads();   // h_t visible before next step's reads
  }
}

// ---------------- small helpers ----------------
__global__ void f32_to_bf16_kernel(const float* __restrict__ in,
                                   __bf16* __restrict__ out, size_t n) {
  size_t i = (size_t)blockIdx.x * blockDim.x + threadIdx.x;
  if (i < n) out[i] = (__bf16)in[i];
}

__global__ void embed_kernel(const int* __restrict__ ids,
                             const float* __restrict__ emb,
                             __bf16* __restrict__ x, size_t n) {
  size_t i = (size_t)blockIdx.x * blockDim.x + threadIdx.x;  // over B*T*EMB
  if (i >= n) return;
  size_t row = i >> 10;                  // EMB = 1024
  int    e   = (int)(i & 1023);
  x[i] = (__bf16)emb[(size_t)ids[row] * EMB + e];
}

// ---------------- launch ----------------
extern "C" void kernel_launch(void* const* d_in, const int* in_sizes, int n_in,
                              void* d_out, int out_size, void* d_ws, size_t ws_size,
                              hipStream_t stream)
{
  (void)in_sizes; (void)n_in; (void)out_size; (void)ws_size;
  const int*   ids  = (const int*)  d_in[0];
  const float* emb  = (const float*)d_in[1];
  const float* Wih0 = (const float*)d_in[2];
  const float* Whh0 = (const float*)d_in[3];
  const float* bih0 = (const float*)d_in[4];
  const float* bhh0 = (const float*)d_in[5];
  const float* Wih1 = (const float*)d_in[6];
  const float* Whh1 = (const float*)d_in[7];
  const float* bih1 = (const float*)d_in[8];
  const float* bhh1 = (const float*)d_in[9];
  const float* fcw  = (const float*)d_in[10];
  const float* fcb  = (const float*)d_in[11];
  float* out = (float*)d_out;

  // workspace carve-up (~141 MB)
  char* ws = (char*)d_ws;
  size_t off = 0;
  auto carve = [&](size_t bytes) -> void* {
    void* p = ws + off;
    off += (bytes + 255) & ~(size_t)255;
    return p;
  };
  __bf16* xbuf  = (__bf16*)carve((size_t)ROWS * HID * 2);   // x, reused as y0
  float*  pbuf  = (float*) carve((size_t)ROWS * HID * 4);   // pre0, reused as pre1
  __bf16* ybuf  = (__bf16*)carve((size_t)ROWS * HID * 2);   // y1
  __bf16* wih0b = (__bf16*)carve((size_t)HID * EMB * 2);
  __bf16* whh0b = (__bf16*)carve((size_t)HID * HID * 2);
  __bf16* wih1b = (__bf16*)carve((size_t)HID * HID * 2);
  __bf16* whh1b = (__bf16*)carve((size_t)HID * HID * 2);
  __bf16* fcwb  = (__bf16*)carve((size_t)VOCAB * HID * 2);

  auto cvt = [&](const float* src, __bf16* dst, size_t n) {
    f32_to_bf16_kernel<<<(unsigned)((n + 255) / 256), 256, 0, stream>>>(src, dst, n);
  };
  cvt(Wih0, wih0b, (size_t)HID * EMB);
  cvt(Whh0, whh0b, (size_t)HID * HID);
  cvt(Wih1, wih1b, (size_t)HID * HID);
  cvt(Whh1, whh1b, (size_t)HID * HID);
  cvt(fcw,  fcwb,  (size_t)VOCAB * HID);

  {
    size_t n = (size_t)ROWS * EMB;
    embed_kernel<<<(unsigned)((n + 255) / 256), 256, 0, stream>>>(ids, emb, xbuf, n);
  }

  float* hT0 = out + (size_t)ROWS * VOCAB;       // hidden layer 0 [B,H]
  float* hT1 = hT0 + (size_t)BATCH * HID;        // hidden layer 1 [B,H]

  // layer 0: pre0 = x @ Wih0^T + bih0 + bhh0 ; scan -> y0 (into xbuf), hT0
  wmma_gemm_bf16<<<dim3(ROWS / 128, HID / 64), 256, 0, stream>>>(
      xbuf, wih0b, pbuf, bih0, bhh0, ROWS, HID, EMB);
  rnn_scan<<<1, 512, 0, stream>>>(pbuf, whh0b, xbuf, hT0);

  // layer 1: pre1 = y0 @ Wih1^T + bih1 + bhh1 ; scan -> y1, hT1
  wmma_gemm_bf16<<<dim3(ROWS / 128, HID / 64), 256, 0, stream>>>(
      xbuf, wih1b, pbuf, bih1, bhh1, ROWS, HID, HID);
  rnn_scan<<<1, 512, 0, stream>>>(pbuf, whh1b, ybuf, hT1);

  // logits = y1 @ fc_w^T + fc_b  (written straight into d_out)
  wmma_gemm_bf16<<<dim3(ROWS / 128, VOCAB / 64), 256, 0, stream>>>(
      ybuf, fcwb, out, fcb, nullptr, ROWS, VOCAB, HID);
}
